// BahdanauAttention_10849087389795
// MI455X (gfx1250) — matrix operand order repaired and checked by reading, not compiled, not run on hardware
//
#include <hip/hip_runtime.h>
#include <hip/hip_bf16.h>
#include <math.h>

// ---------------------------------------------------------------------------
// Bahdanau attention for MI455X (gfx1250, wave32, WMMA), double-buffered.
//   h     = tanh(Q*W1^T + b1 + V*W2^T + b2)      (GEMM, fused 2-input)
//   score = h*Vw^T + vb                          (GEMM)
//   attn  = softmax(score, axis=-1)              (row softmax, 1024)
//   ctx   = attn @ value (batched)               (GEMM, B K-major)
// fp32 in HBM; tiles converted to bf16 into double-buffered LDS with
// register prefetch of the next K-tile, so HBM latency overlaps the
// v_wmma_f32_16x16x32_bf16 stream. d_out = [attn | ctx]; ctx half doubles
// as scratch for h. No d_ws usage.
// ---------------------------------------------------------------------------

typedef __bf16 bf16_t;
typedef __attribute__((ext_vector_type(16))) __bf16 v16bf;
typedef __attribute__((ext_vector_type(8)))  __bf16 v8bf;
typedef __attribute__((ext_vector_type(8)))  float  v8f;

#define BM 128
#define BN 128
#define KT 64          // K-tile per LDS stage (two 16x16x32 K-steps)
#define NTHREADS 256

// ---- fragment loads, per CDNA5 ISA 16-bit WMMA VGPR layouts -----------------
// A (16x32 bf16): lanes 0-15 hold row M=lane, K chunks [0..7],[16..23];
//                 lanes 16-31 hold row M=lane-16, K chunks [8..15],[24..31].
__device__ __forceinline__ v16bf load_a_frag(const bf16_t* ldsA, int rowbase,
                                             int lane, int kstep) {
    const int r  = rowbase + (lane & 15);
    const int kh = (lane >> 4) << 3;          // 0 or 8
    const bf16_t* p = ldsA + r * KT + kstep * 32;
    v8bf lo = *(const v8bf*)(p + kh);         // 16B ds_load
    v8bf hi = *(const v8bf*)(p + 16 + kh);    // 16B ds_load
    v16bf a;
#pragma unroll
    for (int i = 0; i < 8; ++i) { a[i] = lo[i]; a[i + 8] = hi[i]; }
    return a;
}
// B (32x16 bf16): lanes 0-15 hold col N=lane, K chunks [0..7],[16..23];
// lanes 16-31 col N=lane-16, K chunks [8..15],[24..31]. LDS B tile stored [n][KT].
__device__ __forceinline__ v16bf load_b_frag(const bf16_t* ldsB, int colbase,
                                             int lane, int kstep) {
    const int c  = colbase + (lane & 15);
    const int kh = (lane >> 4) << 3;          // 0 or 8
    const bf16_t* p = ldsB + c * KT + kstep * 32;
    v8bf lo = *(const v8bf*)(p + kh);
    v8bf hi = *(const v8bf*)(p + 16 + kh);
    v16bf b;
#pragma unroll
    for (int i = 0; i < 8; ++i) { b[i] = lo[i]; b[i + 8] = hi[i]; }
    return b;
}

// ---- global -> register prefetch of one 128xKT (and KTx128) tile ------------
template <bool BKXN>
__device__ __forceinline__ void stage_to_regs(const float* __restrict__ A,
                                              const float* __restrict__ Bsrc,
                                              int lda, int ldb,
                                              int block_m, int block_n, int k0,
                                              float4 pa[8], float4 pb[8]) {
#pragma unroll
    for (int i = 0; i < 8; ++i) {             // A tile: 128 rows x 64 k
        const int linear = threadIdx.x + i * NTHREADS;  // 0..2047 float4 slots
        const int row = linear >> 4;                    // 16 float4 per row
        const int kq  = (linear & 15) << 2;
        pa[i] = *(const float4*)(A + (size_t)(block_m + row) * lda + k0 + kq);
    }
    if (!BKXN) {                              // B = weights, [N][K] layout
#pragma unroll
        for (int i = 0; i < 8; ++i) {
            const int linear = threadIdx.x + i * NTHREADS;
            const int n  = linear >> 4;
            const int kq = (linear & 15) << 2;
            pb[i] = *(const float4*)(Bsrc + (size_t)(block_n + n) * ldb + k0 + kq);
        }
    } else {                                  // B = value, [K][N] layout
#pragma unroll
        for (int i = 0; i < 8; ++i) {
            const int linear = threadIdx.x + i * NTHREADS;
            const int k  = linear >> 5;                 // 32 float4 per k row
            const int nq = (linear & 31) << 2;
            pb[i] = *(const float4*)(Bsrc + (size_t)(k0 + k) * ldb + block_n + nq);
        }
    }
}

// ---- register -> LDS (fp32 -> bf16 convert on store) ------------------------
template <bool BKXN>
__device__ __forceinline__ void regs_to_lds(const float4 pa[8], const float4 pb[8],
                                            bf16_t* ldsA, bf16_t* ldsB) {
#pragma unroll
    for (int i = 0; i < 8; ++i) {
        const int linear = threadIdx.x + i * NTHREADS;
        const int row = linear >> 4;
        const int kq  = (linear & 15) << 2;
        union { bf16_t h[4]; unsigned long long u; } u;
        u.h[0] = (bf16_t)pa[i].x; u.h[1] = (bf16_t)pa[i].y;
        u.h[2] = (bf16_t)pa[i].z; u.h[3] = (bf16_t)pa[i].w;
        *(unsigned long long*)&ldsA[row * KT + kq] = u.u;
    }
    if (!BKXN) {
#pragma unroll
        for (int i = 0; i < 8; ++i) {
            const int linear = threadIdx.x + i * NTHREADS;
            const int n  = linear >> 4;
            const int kq = (linear & 15) << 2;
            union { bf16_t h[4]; unsigned long long u; } u;
            u.h[0] = (bf16_t)pb[i].x; u.h[1] = (bf16_t)pb[i].y;
            u.h[2] = (bf16_t)pb[i].z; u.h[3] = (bf16_t)pb[i].w;
            *(unsigned long long*)&ldsB[n * KT + kq] = u.u;
        }
    } else {                                  // transpose into [n][KT]
#pragma unroll
        for (int i = 0; i < 8; ++i) {
            const int linear = threadIdx.x + i * NTHREADS;
            const int k  = linear >> 5;
            const int nq = (linear & 31) << 2;
            ldsB[(nq + 0) * KT + k] = (bf16_t)pb[i].x;
            ldsB[(nq + 1) * KT + k] = (bf16_t)pb[i].y;
            ldsB[(nq + 2) * KT + k] = (bf16_t)pb[i].z;
            ldsB[(nq + 3) * KT + k] = (bf16_t)pb[i].w;
        }
    }
}

// ---- one K-sweep: acc += A(block_m..+127,:) * B(block_n..+127,:)^T ----------
// Double-buffered LDS, register-prefetch pipeline (one barrier per K-tile).
template <bool BKXN>
__device__ __forceinline__ void mma_pass(const float* __restrict__ A,
                                         const float* __restrict__ Bsrc,
                                         int K, int lda, int ldb,
                                         int block_m, int block_n,
                                         bf16_t* ldsA, bf16_t* ldsB,
                                         v8f acc[4][2]) {
    const int lane = threadIdx.x & 31;
    const int wave = threadIdx.x >> 5;
    const int wm   = wave & 1;                // 0..1  -> 64-row band
    const int wn   = wave >> 1;               // 0..3  -> 32-col band
    const int nk   = K / KT;

    float4 pa[8], pb[8];
    stage_to_regs<BKXN>(A, Bsrc, lda, ldb, block_m, block_n, 0, pa, pb);
    regs_to_lds<BKXN>(pa, pb, ldsA, ldsB);    // fill buffer 0

    for (int kt = 0; kt < nk; ++kt) {
        const int cur  = kt & 1;
        const int nxt  = cur ^ 1;
        const bool more = (kt + 1 < nk);
        if (more)                             // issue next tile's HBM loads now
            stage_to_regs<BKXN>(A, Bsrc, lda, ldb, block_m, block_n,
                                (kt + 1) * KT, pa, pb);
        __syncthreads();                      // buffer `cur` ready for all waves

        const bf16_t* cA = ldsA + cur * (BM * KT);
        const bf16_t* cB = ldsB + cur * (BN * KT);
#pragma unroll
        for (int s = 0; s < 2; ++s) {         // two 16x16x32 K-steps
            v16bf af[4], bfg[2];
#pragma unroll
            for (int mt = 0; mt < 4; ++mt)
                af[mt] = load_a_frag(cA, wm * 64 + mt * 16, lane, s);
#pragma unroll
            for (int nt = 0; nt < 2; ++nt)
                bfg[nt] = load_b_frag(cB, wn * 32 + nt * 16, lane, s);
#pragma unroll
            for (int mt = 0; mt < 4; ++mt)
#pragma unroll
                for (int nt = 0; nt < 2; ++nt)
                    acc[mt][nt] = __builtin_amdgcn_wmma_f32_16x16x32_bf16(
                        false, af[mt], false, bfg[nt], (short)0,
                        acc[mt][nt], false, false);
        }
        if (more)                             // drain prefetch into back buffer
            regs_to_lds<BKXN>(pa, pb, ldsA + nxt * (BM * KT),
                              ldsB + nxt * (BN * KT));
    }
    __syncthreads();                          // protect buffers for a next pass
}

// ---- generic tiled GEMM: C = act( A*B^T (+ A1*B1^T) (+bias) ) --------------
template <bool TWO_IN, bool TANH_ACT, bool BKXN, bool HAS_BIAS>
__global__ __launch_bounds__(NTHREADS)
void gemm_wmma_bf16(const float* __restrict__ A0, const float* __restrict__ B0,
                    const float* __restrict__ A1, const float* __restrict__ B1,
                    const float* __restrict__ bias0, const float* __restrict__ bias1,
                    float* __restrict__ C,
                    int K, int lda, int ldb, int ldc,
                    size_t strideA, size_t strideB, size_t strideC) {
    __shared__ bf16_t ldsA[2 * BM * KT];      // 32 KB
    __shared__ bf16_t ldsB[2 * BN * KT];      // 32 KB

    const int block_m = blockIdx.x * BM;
    const int block_n = blockIdx.y * BN;
    const size_t zb = blockIdx.z;

    v8f acc[4][2];
#pragma unroll
    for (int mt = 0; mt < 4; ++mt)
#pragma unroll
        for (int nt = 0; nt < 2; ++nt) acc[mt][nt] = (v8f)(0.0f);

    mma_pass<BKXN>(A0 + zb * strideA, B0 + zb * strideB, K, lda, ldb,
                   block_m, block_n, ldsA, ldsB, acc);
    if (TWO_IN)
        mma_pass<BKXN>(A1 + zb * strideA, B1 + zb * strideB, K, lda, ldb,
                       block_m, block_n, ldsA, ldsB, acc);

    // epilogue per C/D layout: VGPR i -> M = (lane>=16 ? 8 : 0)+i, N = lane&15
    const int lane = threadIdx.x & 31;
    const int wave = threadIdx.x >> 5;
    const int wm = wave & 1, wn = wave >> 1;
    const int cn = lane & 15;
    const int mh = (lane >> 4) * 8;
    float* Cb = C + zb * strideC;
#pragma unroll
    for (int mt = 0; mt < 4; ++mt) {
#pragma unroll
        for (int nt = 0; nt < 2; ++nt) {
            const int col = block_n + wn * 32 + nt * 16 + cn;
            float badd = 0.0f;
            if (HAS_BIAS) { badd = bias0[col]; if (TWO_IN) badd += bias1[col]; }
            const size_t rowbase = (size_t)(block_m + wm * 64 + mt * 16 + mh);
#pragma unroll
            for (int i = 0; i < 8; ++i) {
                float v = acc[mt][nt][i] + badd;
                if (TANH_ACT) v = tanhf(v);
                Cb[(rowbase + i) * (size_t)ldc + col] = v;
            }
        }
    }
}

// ---- in-place row softmax over 1024 columns (one block per row) -------------
__global__ __launch_bounds__(NTHREADS)
void softmax1024(float* __restrict__ x) {
    const size_t row = blockIdx.x;
    float* p = x + row * 1024;
    const int t = threadIdx.x;
    __shared__ float red[8];

    float v[4];
    float m = -3.4e38f;
#pragma unroll
    for (int i = 0; i < 4; ++i) { v[i] = p[t + 256 * i]; m = fmaxf(m, v[i]); }
#pragma unroll
    for (int off = 16; off > 0; off >>= 1) m = fmaxf(m, __shfl_xor(m, off, 32));
    if ((t & 31) == 0) red[t >> 5] = m;
    __syncthreads();
#pragma unroll
    for (int j = 0; j < 8; ++j) m = fmaxf(m, red[j]);

    float s = 0.0f;
#pragma unroll
    for (int i = 0; i < 4; ++i) { v[i] = __expf(v[i] - m); s += v[i]; }
#pragma unroll
    for (int off = 16; off > 0; off >>= 1) s += __shfl_xor(s, off, 32);
    __syncthreads();
    if ((t & 31) == 0) red[t >> 5] = s;
    __syncthreads();
    s = 0.0f;
#pragma unroll
    for (int j = 0; j < 8; ++j) s += red[j];
    const float inv = 1.0f / s;
#pragma unroll
    for (int i = 0; i < 4; ++i) p[t + 256 * i] = v[i] * inv;
}

extern "C" void kernel_launch(void* const* d_in, const int* in_sizes, int n_in,
                              void* d_out, int out_size, void* d_ws, size_t ws_size,
                              hipStream_t stream) {
    (void)n_in; (void)d_ws; (void)ws_size; (void)out_size;
    const float* query = (const float*)d_in[0];
    // d_in[1] = key (unused by reference)
    const float* value = (const float*)d_in[2];
    const float* w1_w  = (const float*)d_in[3];
    const float* w1_b  = (const float*)d_in[4];
    const float* w2_w  = (const float*)d_in[5];
    const float* w2_b  = (const float*)d_in[6];
    const float* v_w   = (const float*)d_in[7];
    const float* v_b   = (const float*)d_in[8];

    const int H  = in_sizes[4];               // 1024
    const int L  = H;                         // kv_length == hidden_size
    const int BL = in_sizes[0] / H;           // B*L = 32768
    const int Bb = BL / L;                    // 32

    float* attn = (float*)d_out;              // B*L*H (output 0)
    float* ctx  = attn + (size_t)BL * H;      // B*L*H (output 1; scratch for h)

    dim3 blk(NTHREADS);

    // 1) h = tanh(Q*W1^T + V*W2^T + b1 + b2) -> ctx half (scratch)
    dim3 g1(BL / BM, H / BN, 1);
    gemm_wmma_bf16<true, true, false, true><<<g1, blk, 0, stream>>>(
        query, w1_w, value, w2_w, w1_b, w2_b, ctx,
        H, H, H, H, (size_t)0, (size_t)0, (size_t)0);

    // 2) score = h*Vw^T + vb -> attn half
    gemm_wmma_bf16<false, false, false, true><<<g1, blk, 0, stream>>>(
        ctx, v_w, nullptr, nullptr, v_b, nullptr, attn,
        H, H, H, H, (size_t)0, (size_t)0, (size_t)0);

    // 3) softmax rows of attn (in place) -> final attention_weights
    softmax1024<<<dim3(BL), blk, 0, stream>>>(attn);

    // 4) ctx = attn @ value (batched; value is K-major) -> overwrite scratch half
    dim3 g4(L / BM, H / BN, Bb);
    gemm_wmma_bf16<false, false, true, false><<<g4, blk, 0, stream>>>(
        attn, value, nullptr, nullptr, nullptr, nullptr, ctx,
        L, H, H, H, (size_t)L * H, (size_t)L * H, (size_t)L * H);
}
